// EncoderLayer_48782238548393
// MI455X (gfx1250) — matrix operand order repaired and checked by reading, not compiled, not run on hardware
//
#include <hip/hip_runtime.h>
#include <math.h>

typedef __attribute__((ext_vector_type(16))) _Float16 v16h;
typedef __attribute__((ext_vector_type(8)))  _Float16 v8h;
typedef __attribute__((ext_vector_type(8)))  float    v8f;

// ---------------------------------------------------------------------------
// Fragment loaders for v_wmma_f32_16x16x32_f16 (wave32), contiguous K order, same for A and B (not the ISA table order)
// A (16x32, MxK) 16-bit: lanes 0-15 hold M=lane, K=0..15 (contig);
//                        lanes 16-31 hold M=lane-16, K=16..31.
// B (32x16, KxN) 16-bit: lanes 0-15 hold N=lane,  K=0..15 (contig);
//                        lanes 16-31 hold N=lane-16, K=16..31.
// C/D (16x16 f32): lanes 0-15: N=lane, vgpr r = row r; lanes 16-31: row r+8.
// All buffers are f16-native: A and W-row B fragments use 16B vector loads; frag_b_knh gathers per element.
// ---------------------------------------------------------------------------

__device__ __forceinline__ v16h frag_a_h(const _Float16* __restrict__ p,
                                         int ld, int row0, int k0, int lane) {
  const _Float16* r = p + (size_t)(row0 + (lane & 15)) * ld + k0 + ((lane >> 4) * 16);
  v8h lo = *(const v8h*)r;          // K octet +0
  v8h hi = *(const v8h*)(r + 8);    // K octet +8
  v16h f;
#pragma unroll
  for (int i = 0; i < 8; ++i) { f[i] = lo[i]; f[i + 8] = hi[i]; }
  return f;
}

// B as transpose of row-major weight W[N,K]: column n of B == row n of W.
__device__ __forceinline__ v16h frag_b_wh(const _Float16* __restrict__ w,
                                          int ld, int col0, int k0, int lane) {
  const _Float16* r = w + (size_t)(col0 + (lane & 15)) * ld + k0 + ((lane >> 4) * 16);
  v8h lo = *(const v8h*)r;
  v8h hi = *(const v8h*)(r + 8);
  v16h f;
#pragma unroll
  for (int i = 0; i < 8; ++i) { f[i] = lo[i]; f[i + 8] = hi[i]; }
  return f;
}

// B directly from a row-major KxN f16 matrix (strided gather down a column).
__device__ __forceinline__ v16h frag_b_knh(const _Float16* __restrict__ bm,
                                           int ld, int col0, int k0, int lane) {
  const _Float16* r = bm + (size_t)(k0 + ((lane >> 4) * 16)) * ld + col0 + (lane & 15);
  v16h f;
#pragma unroll
  for (int i = 0; i < 16; ++i) f[i] = r[(size_t)i * ld];
  return f;
}

// ---------------------------------------------------------------------------
// Generic GEMM: C[M,N] = act(A[M,K] @ W[N,K]^T + bias),  A/W f16, acc f32.
// out_f16: store f16 (feeds another GEMM) else f32 (feeds scalar math).
// blockdiag: per-head 64x64 weight (Ww/Wu); A K-window follows output head.
// Block = 256 threads = 8 waves (2 M x 4 N), wave tile 64x64, block 128x256.
// ---------------------------------------------------------------------------
__global__ void __launch_bounds__(256)
gemm_h(const _Float16* __restrict__ A, const _Float16* __restrict__ W,
       const float* __restrict__ bias, float* __restrict__ Cf,
       _Float16* __restrict__ Ch, int M, int N, int K, int lda,
       int relu, int blockdiag, int out_f16) {
  const int lane = threadIdx.x & 31;
  const int wave = threadIdx.x >> 5;
  const int wm = wave >> 2;
  const int wn = wave & 3;
  const int row0 = blockIdx.y * 128 + wm * 64;
  const int col0 = blockIdx.x * 256 + wn * 64;

  const int aKoff = blockdiag ? col0 : 0;
  const int wrow0 = blockdiag ? 0 : col0;

  v8f acc[4][4] = {};

  for (int k0 = 0; k0 < K; k0 += 32) {
    // prefetch next K tile of A (lowers to global_prefetch_b8)
    if (k0 + 32 < K)
      __builtin_prefetch(A + (size_t)(row0 + (lane & 15)) * lda + aKoff + k0 + 32, 0, 3);
    v16h af[4], bf[4];
#pragma unroll
    for (int mi = 0; mi < 4; ++mi)
      af[mi] = frag_a_h(A, lda, row0 + mi * 16, aKoff + k0, lane);
#pragma unroll
    for (int ni = 0; ni < 4; ++ni)
      bf[ni] = frag_b_wh(W, K, wrow0 + ni * 16, k0, lane);
#pragma unroll
    for (int mi = 0; mi < 4; ++mi)
#pragma unroll
      for (int ni = 0; ni < 4; ++ni)
        acc[mi][ni] = __builtin_amdgcn_wmma_f32_16x16x32_f16(
            false, af[mi], false, bf[ni], (short)0, acc[mi][ni], false, false);
  }

  const int nl = lane & 15;
  const int mh = (lane >> 4) * 8;
#pragma unroll
  for (int mi = 0; mi < 4; ++mi) {
#pragma unroll
    for (int ni = 0; ni < 4; ++ni) {
      const int col = col0 + ni * 16 + nl;
      const float bv = bias[blockdiag ? (col & 63) : col];
#pragma unroll
      for (int r = 0; r < 8; ++r) {
        const int row = row0 + mi * 16 + mh + r;
        float v = acc[mi][ni][r] + bv;
        if (relu && v < 0.0f) v = 0.0f;
        if (out_f16) Ch[(size_t)row * N + col] = (_Float16)v;
        else         Cf[(size_t)row * N + col] = v;
      }
    }
  }
}

// ---------------------------------------------------------------------------
// AV: per (b,h): x[l, h*64+d] = sum_k att[b,h,l,k] * Vlin[b*512+k, h*64+d]
// att and V are f16; output f16 (consumed by Wo GEMM).
// ---------------------------------------------------------------------------
__global__ void __launch_bounds__(256)
av_h(const _Float16* __restrict__ att, const _Float16* __restrict__ Vlin,
     _Float16* __restrict__ xout) {
  const int bh = blockIdx.x;           // 0..15
  const int b = bh >> 3, h = bh & 7;
  const int lane = threadIdx.x & 31;
  const int wave = threadIdx.x >> 5;
  const int row0 = wave * 64;

  const _Float16* Ab = att + (size_t)bh * 512 * 512;
  const _Float16* Bb = Vlin + (size_t)b * 512 * 512 + h * 64;

  v8f acc[4][4] = {};
  for (int k0 = 0; k0 < 512; k0 += 32) {
    v16h af[4], bf[4];
#pragma unroll
    for (int mi = 0; mi < 4; ++mi)
      af[mi] = frag_a_h(Ab, 512, row0 + mi * 16, k0, lane);
#pragma unroll
    for (int ni = 0; ni < 4; ++ni)
      bf[ni] = frag_b_knh(Bb, 512, ni * 16, k0, lane);
#pragma unroll
    for (int mi = 0; mi < 4; ++mi)
#pragma unroll
      for (int ni = 0; ni < 4; ++ni)
        acc[mi][ni] = __builtin_amdgcn_wmma_f32_16x16x32_f16(
            false, af[mi], false, bf[ni], (short)0, acc[mi][ni], false, false);
  }

  const int nl = lane & 15;
  const int mh = (lane >> 4) * 8;
#pragma unroll
  for (int mi = 0; mi < 4; ++mi)
#pragma unroll
    for (int ni = 0; ni < 4; ++ni)
#pragma unroll
      for (int r = 0; r < 8; ++r) {
        const int row = row0 + mi * 16 + mh + r;
        xout[((size_t)(b * 512 + row)) * 512 + h * 64 + ni * 16 + nl] =
            (_Float16)acc[mi][ni][r];
      }
}

// ---------------------------------------------------------------------------
// Additive-attention energy + softmax: one block per (b,h,q) row.
// e[k] = vb + sum_d vw[d]*tanh(Qp[q,d]+Kp[k,d]); masked softmax -> f16 probs.
// ---------------------------------------------------------------------------
__global__ void __launch_bounds__(256)
energy_softmax(const float* __restrict__ Qp, const float* __restrict__ Kp,
               const int* __restrict__ mask, const float* __restrict__ vw,
               const float* __restrict__ vb, _Float16* __restrict__ att) {
  const int idx = blockIdx.x;           // ((b*8+h)*512+q)
  const int b = idx >> 12;
  const int h = (idx >> 9) & 7;
  const int q = idx & 511;
  const int tid = threadIdx.x;

  __shared__ float qrow[64];
  __shared__ float red[256];

  if (tid < 64)
    qrow[tid] = Qp[((size_t)(b * 512 + q)) * 512 + h * 64 + tid];
  __syncthreads();

  const float vbv = vb[0];
  float e[2];
#pragma unroll
  for (int j = 0; j < 2; ++j) {
    const int k = tid + j * 256;
    const float* kp = Kp + ((size_t)(b * 512 + k)) * 512 + h * 64;
    float s = 0.0f;
#pragma unroll 8
    for (int d = 0; d < 64; ++d) s += vw[d] * tanhf(qrow[d] + kp[d]);
    s += vbv;
    if (mask[b * 512 + k] == 0) s = -INFINITY;
    e[j] = s;
  }

  red[tid] = fmaxf(e[0], e[1]);
  __syncthreads();
  for (int s = 128; s > 0; s >>= 1) {
    if (tid < s) red[tid] = fmaxf(red[tid], red[tid + s]);
    __syncthreads();
  }
  const float m = red[0];
  __syncthreads();

  const float p0 = __expf(e[0] - m);
  const float p1 = __expf(e[1] - m);
  red[tid] = p0 + p1;
  __syncthreads();
  for (int s = 128; s > 0; s >>= 1) {
    if (tid < s) red[tid] += red[tid + s];
    __syncthreads();
  }
  const float inv = 1.0f / red[0];

  const size_t obase = (size_t)idx * 512;
  att[obase + tid]       = (_Float16)(p0 * inv);
  att[obase + tid + 256] = (_Float16)(p1 * inv);
}

// ---------------------------------------------------------------------------
// out = LayerNorm(res + x) * g + b  (row length 512); optional f16 mirror.
// ---------------------------------------------------------------------------
__global__ void __launch_bounds__(256)
residual_ln(const float* __restrict__ res, const float* __restrict__ x,
            const float* __restrict__ g, const float* __restrict__ bta,
            float* __restrict__ outf, _Float16* __restrict__ outh) {
  const int row = blockIdx.x;
  const int tid = threadIdx.x;
  __shared__ float s1[256], s2[256];

  const size_t base = (size_t)row * 512;
  const float a0 = res[base + tid] + x[base + tid];
  const float a1 = res[base + tid + 256] + x[base + tid + 256];
  s1[tid] = a0 + a1;
  s2[tid] = a0 * a0 + a1 * a1;
  __syncthreads();
  for (int s = 128; s > 0; s >>= 1) {
    if (tid < s) { s1[tid] += s1[tid + s]; s2[tid] += s2[tid + s]; }
    __syncthreads();
  }
  const float mu = s1[0] * (1.0f / 512.0f);
  const float var = s2[0] * (1.0f / 512.0f) - mu * mu;
  const float inv = rsqrtf(var + 1e-5f);
  const float y0 = (a0 - mu) * inv * g[tid] + bta[tid];
  const float y1 = (a1 - mu) * inv * g[tid + 256] + bta[tid + 256];
  outf[base + tid] = y0;
  outf[base + tid + 256] = y1;
  if (outh) {
    outh[base + tid] = (_Float16)y0;
    outh[base + tid + 256] = (_Float16)y1;
  }
}

// One-shot fp32 -> f16 conversion (weights / src), amortized over all tiles.
__global__ void __launch_bounds__(256)
cvt_f16(const float* __restrict__ in, _Float16* __restrict__ out, int n) {
  const int i = blockIdx.x * 256 + threadIdx.x;
  if (i < n) out[i] = (_Float16)in[i];
}

// ---------------------------------------------------------------------------
extern "C" void kernel_launch(void* const* d_in, const int* in_sizes, int n_in,
                              void* d_out, int out_size, void* d_ws, size_t ws_size,
                              hipStream_t stream) {
  const float* src   = (const float*)d_in[0];
  const int*   smask = (const int*)  d_in[1];
  const float* Wq = (const float*)d_in[2],  *bq = (const float*)d_in[3];
  const float* Wk = (const float*)d_in[4],  *bk = (const float*)d_in[5];
  const float* Wv = (const float*)d_in[6],  *bv = (const float*)d_in[7];
  const float* Ww = (const float*)d_in[8],  *bw = (const float*)d_in[9];
  const float* Wu = (const float*)d_in[10], *bu = (const float*)d_in[11];
  const float* vw = (const float*)d_in[12], *vb = (const float*)d_in[13];
  const float* Wo = (const float*)d_in[14], *bo = (const float*)d_in[15];
  const float* ln1g = (const float*)d_in[16], *ln1b = (const float*)d_in[17];
  const float* ln2g = (const float*)d_in[18], *ln2b = (const float*)d_in[19];
  const float* W1 = (const float*)d_in[20], *b1 = (const float*)d_in[21];
  const float* W2 = (const float*)d_in[22], *b2 = (const float*)d_in[23];
  float* out = (float*)d_out;

  const int M = 1024, HID = 512, PF = 2048;

  // bump allocator over workspace (256B aligned slices)
  char* wp = (char*)d_ws;
  auto alloc_h = [&](size_t n) {
    _Float16* r = (_Float16*)wp; wp += ((n * 2 + 255) & ~(size_t)255); return r; };
  auto alloc_f = [&](size_t n) {
    float* r = (float*)wp; wp += ((n * 4 + 255) & ~(size_t)255); return r; };

  _Float16* hsrc = alloc_h((size_t)M * HID);
  _Float16* hWq = alloc_h((size_t)HID * HID);
  _Float16* hWk = alloc_h((size_t)HID * HID);
  _Float16* hWv = alloc_h((size_t)HID * HID);
  _Float16* hWo = alloc_h((size_t)HID * HID);
  _Float16* hW1 = alloc_h((size_t)PF * HID);
  _Float16* hW2 = alloc_h((size_t)HID * PF);
  _Float16* hWw = alloc_h(64 * 64);
  _Float16* hWu = alloc_h(64 * 64);
  _Float16* hQ  = alloc_h((size_t)M * HID);
  _Float16* hK  = alloc_h((size_t)M * HID);
  _Float16* hV  = alloc_h((size_t)M * HID);
  float*    Qp  = alloc_f((size_t)M * HID);
  float*    Kp  = alloc_f((size_t)M * HID);
  _Float16* hatt = alloc_h((size_t)2 * 8 * 512 * 512);
  _Float16* hxatt = alloc_h((size_t)M * HID);
  float*    xo  = alloc_f((size_t)M * HID);
  float*    s1f = alloc_f((size_t)M * HID);
  _Float16* s1h = alloc_h((size_t)M * HID);
  _Float16* hff = alloc_h((size_t)M * PF);
  float*    x2  = alloc_f((size_t)M * HID);

  dim3 blk(256, 1, 1);
  auto cvt = [&](const float* in, _Float16* o, int n) {
    cvt_f16<<<dim3((n + 255) / 256), blk, 0, stream>>>(in, o, n);
  };
  auto gemm = [&](const _Float16* A, const _Float16* W, const float* bias,
                  float* Cf, _Float16* Ch, int m, int n, int k, int lda,
                  int relu, int bd, int of16) {
    dim3 grid(n / 256, m / 128);
    gemm_h<<<grid, blk, 0, stream>>>(A, W, bias, Cf, Ch, m, n, k, lda, relu, bd, of16);
  };

  // one-shot f16 conversion of src + all weights
  cvt(src, hsrc, M * HID);
  cvt(Wq, hWq, HID * HID);  cvt(Wk, hWk, HID * HID);
  cvt(Wv, hWv, HID * HID);  cvt(Wo, hWo, HID * HID);
  cvt(W1, hW1, PF * HID);   cvt(W2, hW2, HID * PF);
  cvt(Ww, hWw, 64 * 64);    cvt(Wu, hWu, 64 * 64);

  // QKV projections (f16 out -> consumed by GEMMs)
  gemm(hsrc, hWq, bq, nullptr, hQ, M, HID, HID, HID, 0, 0, 1);
  gemm(hsrc, hWk, bk, nullptr, hK, M, HID, HID, HID, 0, 0, 1);
  gemm(hsrc, hWv, bv, nullptr, hV, M, HID, HID, HID, 0, 0, 1);
  // per-head additive-attention projections (f32 out -> tanh energy)
  gemm(hQ, hWw, bw, Qp, nullptr, M, HID, 64, HID, 0, 1, 0);
  gemm(hK, hWu, bu, Kp, nullptr, M, HID, 64, HID, 0, 1, 0);
  // energy + masked softmax -> f16 probabilities
  energy_softmax<<<dim3(2 * 8 * 512), blk, 0, stream>>>(Qp, Kp, smask, vw, vb, hatt);
  // attention @ V (f16 -> f16)
  av_h<<<dim3(16), blk, 0, stream>>>(hatt, hV, hxatt);
  // output projection (f32 out -> LN), residual + LN1 (f32 + f16 mirrors)
  gemm(hxatt, hWo, bo, xo, nullptr, M, HID, HID, HID, 0, 0, 0);
  residual_ln<<<dim3(M), blk, 0, stream>>>(src, xo, ln1g, ln1b, s1f, s1h);
  // FFN: ReLU hidden in f16, second GEMM back to f32 for LN2
  gemm(s1h, hW1, b1, nullptr, hff, M, PF, HID, HID, 1, 0, 1);
  gemm(hff, hW2, b2, x2, nullptr, M, HID, PF, PF, 0, 0, 0);
  residual_ln<<<dim3(M), blk, 0, stream>>>(s1f, x2, ln2g, ln2b, out, nullptr);
}
